// ObjectContextBlock_39324720562795
// MI455X (gfx1250) — matrix operand order repaired and checked by reading, not compiled, not run on hardware
//
#include <hip/hip_runtime.h>

// Problem constants (from reference)
#define BATCH 4
#define CIN   512
#define HW    16384     // 128*128
#define KC    256
#define KP    150
#define KPP   160       // KP padded to multiple of 32 (zero padding is exact)
#define MT    64        // pixel tile per workgroup

typedef __attribute__((ext_vector_type(16))) __bf16 v16bf;
typedef __attribute__((ext_vector_type(8)))  __bf16 v8bf;
typedef __attribute__((ext_vector_type(8)))  float  v8f;

static __device__ __forceinline__ __bf16 f2bf(float f) {
  unsigned u = __float_as_uint(f);
  u += 0x7FFFu + ((u >> 16) & 1u);          // round-to-nearest-even
  unsigned short s = (unsigned short)(u >> 16);
  __bf16 r; __builtin_memcpy(&r, &s, 2); return r;
}

static __device__ __forceinline__ v16bf cat16(v8bf lo, v8bf hi) {
  return __builtin_shufflevector(lo, hi, 0,1,2,3,4,5,6,7,8,9,10,11,12,13,14,15);
}

// A fragment (16x32 bf16) from row-major [rows][rowStride]; rows m0..m0+15, K chunk k0.
// ISA layout: lane L (m=L&15, h=L>>4) holds K = h*8+j (j<8) and 16+h*8+(j-8) (j>=8).
static __device__ __forceinline__ v16bf load_a(const __bf16* base, int rowStride,
                                               int m0, int k0, int lane) {
  const __bf16* row = base + (size_t)(m0 + (lane & 15)) * rowStride + k0 + (lane >> 4) * 8;
  v8bf lo = *(const v8bf*)(row);
  v8bf hi = *(const v8bf*)(row + 16);
  return cat16(lo, hi);
}

// B fragment (32x16 bf16) from N-major K-contiguous [ncols][kld]; cols n0..n0+15, K chunk k0.
// ISA layout: lane L holds column n=L&15, K = (L>>4)*8+j (j<8) and 16+(L>>4)*8+(j-8) (j>=8).
static __device__ __forceinline__ v16bf load_b(const __bf16* base, int kld,
                                               int n0, int k0, int lane) {
  const __bf16* p = base + (size_t)(n0 + (lane & 15)) * kld + k0 + (lane >> 4) * 8;
  v8bf lo = *(const v8bf*)(p);
  v8bf hi = *(const v8bf*)(p + 16);
  return cat16(lo, hi);
}

#define WMMA_BF16(A, Bm, Cm) \
  __builtin_amdgcn_wmma_f32_16x16x32_bf16(false, (A), false, (Bm), (short)0, (Cm), false, false)

// ---------------------------------------------------------------------------
// fp32 -> bf16 convert (weights keep natural [N][K] layout = B-fragment layout)
__global__ __launch_bounds__(256) void k_cvt_bf16(const float* __restrict__ src,
                                                  __bf16* __restrict__ dst, int n) {
  int i = blockIdx.x * 256 + threadIdx.x;
  int stride = gridDim.x * 256;
  for (; i < n; i += stride) dst[i] = f2bf(src[i]);
}

// x [B][C][HW] f32  ->  xa [B*HW][C] bf16 (LDS 32x32 tile transpose)
__global__ __launch_bounds__(256) void k_xpose(const float* __restrict__ x,
                                               __bf16* __restrict__ xa) {
  __shared__ float t[32][33];
  int b = blockIdx.z, c0 = blockIdx.y * 32, p0 = blockIdx.x * 32;
  int tid = threadIdx.x;
#pragma unroll
  for (int i = 0; i < 4; ++i) {
    int idx = tid + i * 256;
    int pl = idx & 31, cl = idx >> 5;
    t[pl][cl] = x[((size_t)(b * CIN + c0 + cl)) * HW + p0 + pl];
  }
  __syncthreads();
#pragma unroll
  for (int i = 0; i < 4; ++i) {
    int idx = tid + i * 256;
    int cl = idx & 31, pl = idx >> 5;
    xa[((size_t)(b * HW + p0 + pl)) * CIN + c0 + cl] = f2bf(t[pl][cl]);
  }
}

// key/value from proxy feats (tiny: ~0.4% of FLOPs, plain fp32)
// keyT [B][KPP][KC] bf16 (zero rows for kp>=150); v [B][KC][KPP] bf16 (zero cols)
__global__ __launch_bounds__(256) void k_kv(const float* __restrict__ proxy,
    const float* __restrict__ wo1, const float* __restrict__ bo1,
    const float* __restrict__ wo2, const float* __restrict__ bo2,
    const float* __restrict__ wd,  const float* __restrict__ bd,
    __bf16* __restrict__ keyT, __bf16* __restrict__ vbf) {
  __shared__ float pf[CIN];
  __shared__ float ho[KC];
  int b = blockIdx.x / KPP, kp = blockIdx.x % KPP;
  int t = threadIdx.x;
  bool valid = (kp < KP);               // block-uniform
  if (valid) {
    pf[t]       = proxy[((size_t)(b * CIN + t)) * KP + kp];
    pf[t + 256] = proxy[((size_t)(b * CIN + t + 256)) * KP + kp];
  }
  __syncthreads();
  float hv = 0.f;
  if (valid) {
    float s = bo1[t];
    for (int c = 0; c < CIN; ++c) s += wo1[t * CIN + c] * pf[c];
    hv = fmaxf(s, 0.f);
  }
  ho[t] = hv;
  __syncthreads();
  float kvout = 0.f, vvout = 0.f;
  if (valid) {
    float sk = bo2[t];
    for (int j = 0; j < KC; ++j) sk += wo2[t * KC + j] * ho[j];
    kvout = fmaxf(sk, 0.f);
    float sv = bd[t];
    for (int c = 0; c < CIN; ++c) sv += wd[t * CIN + c] * pf[c];
    vvout = fmaxf(sv, 0.f);
  }
  keyT[((size_t)(b * KPP + kp)) * KC + t] = f2bf(kvout);
  vbf[((size_t)(b * KC + t)) * KPP + kp] = f2bf(vvout);
}

// q = relu(W_p2 @ relu(W_p1 @ x + b1) + b2), fused, per 64-pixel tile.
// 8 waves: wave w -> row pair (w>>2)*32 (2 mtiles), 64 N-columns = 4 ntiles.
// Per K-chunk: 2 A-loads + 4 B-loads feed 8 WMMAs (0.75 loads/WMMA).
__global__ __launch_bounds__(256) void k_query(const __bf16* __restrict__ xa,
    const __bf16* __restrict__ wp1, const float* __restrict__ bp1,
    const __bf16* __restrict__ wp2, const float* __restrict__ bp2,
    __bf16* __restrict__ q) {
  __shared__ __bf16 ldsX[MT * (CIN + 8)];
  __shared__ __bf16 ldsH[MT * (KC + 8)];
  const int t = threadIdx.x, lane = t & 31, w = t >> 5;
  const size_t base = (size_t)blockIdx.x * MT;

  // stage X tile (64 x 512 bf16 = 64KB), coalesced 16B chunks
#pragma unroll
  for (int i = 0; i < 16; ++i) {
    int cid = t + 256 * i;
    int row = cid >> 6, col = (cid & 63) * 8;
    v8bf v = *(const v8bf*)(xa + (base + row) * CIN + col);
    *(v8bf*)(&ldsX[row * (CIN + 8) + col]) = v;
  }
  __syncthreads();

  const int m0 = (w >> 2) * 32;
  const int n0 = (w & 3) * 64;
  const int hl = lane >> 4, nn = lane & 15;

  // GEMM1: h[64x256] = relu(x @ W_p1^T + b1), K = 512
  {
    v8f acc[2][4] = {};
    for (int kc = 0; kc < CIN / 32; ++kc) {
      v16bf a0 = load_a(ldsX, CIN + 8, m0,      kc * 32, lane);
      v16bf a1 = load_a(ldsX, CIN + 8, m0 + 16, kc * 32, lane);
#pragma unroll
      for (int nt = 0; nt < 4; ++nt) {
        v16bf bm = load_b(wp1, CIN, n0 + nt * 16, kc * 32, lane);
        acc[0][nt] = WMMA_BF16(a0, bm, acc[0][nt]);
        acc[1][nt] = WMMA_BF16(a1, bm, acc[1][nt]);
      }
    }
#pragma unroll
    for (int s = 0; s < 2; ++s)
#pragma unroll
      for (int nt = 0; nt < 4; ++nt) {
        int ncol = n0 + nt * 16 + nn;
        float bias = bp1[ncol];
#pragma unroll
        for (int r = 0; r < 8; ++r) {
          int m = m0 + s * 16 + hl * 8 + r;      // D layout: m = (lane>>4)*8 + r
          ldsH[m * (KC + 8) + ncol] = f2bf(fmaxf(acc[s][nt][r] + bias, 0.f));
        }
      }
  }
  __syncthreads();

  // GEMM2: q[64x256] = relu(h @ W_p2^T + b2), K = 256
  {
    v8f acc[2][4] = {};
    for (int kc = 0; kc < KC / 32; ++kc) {
      v16bf a0 = load_a(ldsH, KC + 8, m0,      kc * 32, lane);
      v16bf a1 = load_a(ldsH, KC + 8, m0 + 16, kc * 32, lane);
#pragma unroll
      for (int nt = 0; nt < 4; ++nt) {
        v16bf bm = load_b(wp2, KC, n0 + nt * 16, kc * 32, lane);
        acc[0][nt] = WMMA_BF16(a0, bm, acc[0][nt]);
        acc[1][nt] = WMMA_BF16(a1, bm, acc[1][nt]);
      }
    }
#pragma unroll
    for (int s = 0; s < 2; ++s)
#pragma unroll
      for (int nt = 0; nt < 4; ++nt) {
        int ncol = n0 + nt * 16 + nn;
        float bias = bp2[ncol];
#pragma unroll
        for (int r = 0; r < 8; ++r) {
          size_t m = base + m0 + s * 16 + hl * 8 + r;
          q[m * KC + ncol] = f2bf(fmaxf(acc[s][nt][r] + bias, 0.f));
        }
      }
  }
}

// fused attention per 64-pixel tile: sim GEMM -> softmax(150) -> ctx GEMM
__global__ __launch_bounds__(256) void k_attn(const __bf16* __restrict__ q,
    const __bf16* __restrict__ keyT, const __bf16* __restrict__ vbf,
    __bf16* __restrict__ ctx) {
  __shared__ __bf16 ldsQ[MT * (KC + 8)];
  __shared__ float  ldsS[MT * (KPP + 8)];
  __shared__ __bf16 ldsP[MT * (KPP + 8)];
  const int t = threadIdx.x, lane = t & 31, w = t >> 5;
  const size_t base = (size_t)blockIdx.x * MT;
  const int b = blockIdx.x / (HW / MT);
  const __bf16* keyB = keyT + (size_t)b * KPP * KC;
  const __bf16* vB   = vbf  + (size_t)b * KC * KPP;

#pragma unroll
  for (int i = 0; i < 8; ++i) {
    int cid = t + 256 * i;
    int row = cid >> 5, col = (cid & 31) * 8;
    v8bf v = *(const v8bf*)(q + (base + row) * KC + col);
    *(v8bf*)(&ldsQ[row * (KC + 8) + col]) = v;
  }
  __syncthreads();

  const int m0 = (w >> 2) * 32;
  const int hl = lane >> 4, nn = lane & 15;

  // sim[64x160] = q @ keyT^T * KC^-0.5 ; K = 256; 10 ntiles over 4 waves (3 each, uniform guard)
  {
    v8f acc[2][3] = {};
    const int ntb = (w & 3) * 3;
    for (int kc = 0; kc < KC / 32; ++kc) {
      v16bf a0 = load_a(ldsQ, KC + 8, m0,      kc * 32, lane);
      v16bf a1 = load_a(ldsQ, KC + 8, m0 + 16, kc * 32, lane);
#pragma unroll
      for (int j = 0; j < 3; ++j) {
        int nt = ntb + j;
        if (nt < KPP / 16) {            // wave-uniform: EXEC stays full
          v16bf bm = load_b(keyB, KC, nt * 16, kc * 32, lane);
          acc[0][j] = WMMA_BF16(a0, bm, acc[0][j]);
          acc[1][j] = WMMA_BF16(a1, bm, acc[1][j]);
        }
      }
    }
#pragma unroll
    for (int s = 0; s < 2; ++s)
#pragma unroll
      for (int j = 0; j < 3; ++j) {
        int nt = ntb + j;
        if (nt < KPP / 16) {
          int ncol = nt * 16 + nn;
#pragma unroll
          for (int r = 0; r < 8; ++r) {
            int m = m0 + s * 16 + hl * 8 + r;
            ldsS[m * (KPP + 8) + ncol] = acc[s][j][r] * 0.0625f;   // 256^-0.5
          }
        }
      }
  }
  __syncthreads();

  // softmax over the valid 150 entries; pad 150..159 with zeros
  if (t < MT) {
    float*  row  = &ldsS[t * (KPP + 8)];
    __bf16* prow = &ldsP[t * (KPP + 8)];
    float mx = -1e30f;
    for (int k2 = 0; k2 < KP; ++k2) mx = fmaxf(mx, row[k2]);
    float sum = 0.f;
    for (int k2 = 0; k2 < KP; ++k2) { float e = __expf(row[k2] - mx); row[k2] = e; sum += e; }
    float inv = 1.f / sum;
    for (int k2 = 0; k2 < KP; ++k2)   prow[k2] = f2bf(row[k2] * inv);
    for (int k2 = KP; k2 < KPP; ++k2) prow[k2] = f2bf(0.f);
  }
  __syncthreads();

  // ctx[64x256] = P @ v ; K = 160 (5 chunks)
  {
    v8f acc[2][4] = {};
    const int n0 = (w & 3) * 64;
    for (int kc = 0; kc < KPP / 32; ++kc) {
      v16bf a0 = load_a(ldsP, KPP + 8, m0,      kc * 32, lane);
      v16bf a1 = load_a(ldsP, KPP + 8, m0 + 16, kc * 32, lane);
#pragma unroll
      for (int nt = 0; nt < 4; ++nt) {
        v16bf bm = load_b(vB, KPP, n0 + nt * 16, kc * 32, lane);
        acc[0][nt] = WMMA_BF16(a0, bm, acc[0][nt]);
        acc[1][nt] = WMMA_BF16(a1, bm, acc[1][nt]);
      }
    }
#pragma unroll
    for (int s = 0; s < 2; ++s)
#pragma unroll
      for (int nt = 0; nt < 4; ++nt) {
        int ncol = (w & 3) * 64 + nt * 16 + nn;
#pragma unroll
        for (int r = 0; r < 8; ++r) {
          size_t m = base + m0 + s * 16 + hl * 8 + r;
          ctx[m * KC + ncol] = f2bf(acc[s][nt][r]);
        }
      }
  }
}

// out[b][c][p] = relu(ctx @ W_u^T + b_u); M = pixels so D-tile rows are
// contiguous pixels per lane -> coalesced float4 stores. Two N-passes of 4
// ntiles each to keep accumulator VGPRs at 64.
__global__ __launch_bounds__(256) void k_out(const __bf16* __restrict__ ctx,
    const __bf16* __restrict__ wu, const float* __restrict__ bu,
    float* __restrict__ out) {
  __shared__ __bf16 ldsC[MT * (KC + 8)];
  const int t = threadIdx.x, lane = t & 31, w = t >> 5;
  const size_t base = (size_t)blockIdx.x * MT;
  const int b = blockIdx.x / (HW / MT);
  const int p0 = (blockIdx.x % (HW / MT)) * MT;

#pragma unroll
  for (int i = 0; i < 8; ++i) {
    int cid = t + 256 * i;
    int row = cid >> 5, col = (cid & 31) * 8;
    v8bf v = *(const v8bf*)(ctx + (base + row) * KC + col);
    *(v8bf*)(&ldsC[row * (KC + 8) + col]) = v;
  }
  __syncthreads();

  const int m0 = (w >> 2) * 32;
  const int hl = lane >> 4, nn = lane & 15;

  for (int half = 0; half < 2; ++half) {
    const int n0 = (w & 3) * 128 + half * 64;   // 4 ntiles per pass
    v8f acc[2][4] = {};
    for (int kc = 0; kc < KC / 32; ++kc) {
      v16bf a0 = load_a(ldsC, KC + 8, m0,      kc * 32, lane);
      v16bf a1 = load_a(ldsC, KC + 8, m0 + 16, kc * 32, lane);
#pragma unroll
      for (int nt = 0; nt < 4; ++nt) {
        v16bf bm = load_b(wu, KC, n0 + nt * 16, kc * 32, lane);
        acc[0][nt] = WMMA_BF16(a0, bm, acc[0][nt]);
        acc[1][nt] = WMMA_BF16(a1, bm, acc[1][nt]);
      }
    }
#pragma unroll
    for (int s = 0; s < 2; ++s)
#pragma unroll
      for (int nt = 0; nt < 4; ++nt) {
        int ch = n0 + nt * 16 + nn;
        float bias = bu[ch];
        int p = p0 + m0 + s * 16 + hl * 8;
        float4 v0, v1;
        v0.x = fmaxf(acc[s][nt][0] + bias, 0.f);
        v0.y = fmaxf(acc[s][nt][1] + bias, 0.f);
        v0.z = fmaxf(acc[s][nt][2] + bias, 0.f);
        v0.w = fmaxf(acc[s][nt][3] + bias, 0.f);
        v1.x = fmaxf(acc[s][nt][4] + bias, 0.f);
        v1.y = fmaxf(acc[s][nt][5] + bias, 0.f);
        v1.z = fmaxf(acc[s][nt][6] + bias, 0.f);
        v1.w = fmaxf(acc[s][nt][7] + bias, 0.f);
        float* o = out + ((size_t)(b * CIN + ch)) * HW + p;
        *(float4*)(o)     = v0;
        *(float4*)(o + 4) = v1;
      }
  }
}

// ---------------------------------------------------------------------------
extern "C" void kernel_launch(void* const* d_in, const int* in_sizes, int n_in,
                              void* d_out, int out_size, void* d_ws, size_t ws_size,
                              hipStream_t stream) {
  (void)in_sizes; (void)n_in; (void)out_size; (void)ws_size;
  const float* x     = (const float*)d_in[0];
  const float* proxy = (const float*)d_in[1];
  const float* w_p1  = (const float*)d_in[2];
  const float* b_p1  = (const float*)d_in[3];
  const float* w_p2  = (const float*)d_in[4];
  const float* b_p2  = (const float*)d_in[5];
  const float* w_o1  = (const float*)d_in[6];
  const float* b_o1  = (const float*)d_in[7];
  const float* w_o2  = (const float*)d_in[8];
  const float* b_o2  = (const float*)d_in[9];
  const float* w_d   = (const float*)d_in[10];
  const float* b_d   = (const float*)d_in[11];
  const float* w_u   = (const float*)d_in[12];
  const float* b_u   = (const float*)d_in[13];
  float* out = (float*)d_out;

  char* ws = (char*)d_ws;
  size_t off = 0;
  auto take = [&](size_t bytes) -> char* {
    char* p = ws + off;
    off = (off + bytes + 255) & ~(size_t)255;
    return p;
  };
  __bf16* xa   = (__bf16*)take((size_t)BATCH * HW * CIN * 2);   // 64 MB
  __bf16* qbf  = (__bf16*)take((size_t)BATCH * HW * KC * 2);    // 32 MB
  __bf16* ctxb = (__bf16*)take((size_t)BATCH * HW * KC * 2);    // 32 MB
  __bf16* wp1b = (__bf16*)take((size_t)KC * CIN * 2);
  __bf16* wp2b = (__bf16*)take((size_t)KC * KC * 2);
  __bf16* wub  = (__bf16*)take((size_t)CIN * KC * 2);
  __bf16* keyT = (__bf16*)take((size_t)BATCH * KPP * KC * 2);
  __bf16* vbf  = (__bf16*)take((size_t)BATCH * KC * KPP * 2);

  k_cvt_bf16<<<256, 256, 0, stream>>>(w_p1, wp1b, KC * CIN);
  k_cvt_bf16<<<128, 256, 0, stream>>>(w_p2, wp2b, KC * KC);
  k_cvt_bf16<<<256, 256, 0, stream>>>(w_u,  wub,  CIN * KC);
  k_xpose<<<dim3(HW / 32, CIN / 32, BATCH), 256, 0, stream>>>(x, xa);
  k_kv<<<BATCH * KPP, 256, 0, stream>>>(proxy, w_o1, b_o1, w_o2, b_o2, w_d, b_d, keyT, vbf);
  k_query<<<BATCH * HW / MT, 256, 0, stream>>>(xa, wp1b, b_p1, wp2b, b_p2, qbf);
  k_attn<<<BATCH * HW / MT, 256, 0, stream>>>(qbf, keyT, vbf, ctxb);
  k_out<<<BATCH * HW / MT, 256, 0, stream>>>(ctxb, wub, b_u, out);
}
